// MambaBlock_32633161515362
// MI455X (gfx1250) — matrix operand order repaired and checked by reading, not compiled, not run on hardware
//
#include <hip/hip_runtime.h>
#include <hip/hip_bf16.h>

typedef __attribute__((ext_vector_type(16))) __bf16 v16bf;
typedef __attribute__((ext_vector_type(8)))  __bf16 v8bf;
typedef __attribute__((ext_vector_type(8)))  float  v8f;

// Problem dims (fixed by the reference)
#define BATCH    2
#define SEQ_L    1024
#define D_MODEL  1024
#define D_INNER  2048
#define D_STATE  16
#define D_CONV   4
#define M_ROWS   (BATCH * SEQ_L)          // 2048 rows for all GEMMs

// ---------------------------------------------------------------------------
// f32 -> bf16 conversion (weights / activations staged once)
// ---------------------------------------------------------------------------
__global__ void f32_to_bf16_kernel(const float* __restrict__ in,
                                   __bf16* __restrict__ out, int n) {
    int i = blockIdx.x * blockDim.x + threadIdx.x;
    if (i < n) out[i] = (__bf16)in[i];
}

// ---------------------------------------------------------------------------
// bf16 WMMA GEMM:  C[M,N] = A[M,K] * W[N,K]^T (+ bias[N])
// Block tile 128x128, BK=64, double-buffered LDS filled by CDNA5 async
// load-to-LDS (ASYNCcnt), so global traffic for K-tile k+1 overlaps the
// 16 v_wmma_f32_16x16x32_bf16 ops on K-tile k. 8 waves (wave32): wave grid
// 4(M) x 2(N); each wave owns a 32x64 tile = 2x4 accumulator tiles.
// ---------------------------------------------------------------------------
__global__ __launch_bounds__(256)
void wmma_gemm_bf16(const __bf16* __restrict__ A, const __bf16* __restrict__ W,
                    const float* __restrict__ bias, float* __restrict__ C,
                    int M, int N, int K) {
    __shared__ __bf16 As[2][128 * 64];   // 2 x 16 KB
    __shared__ __bf16 Bs[2][128 * 64];   // 2 x 16 KB

    const int tid  = threadIdx.x;
    const int lane = tid & 31;
    const int wid  = tid >> 5;
    const int wm   = wid & 3;          // 0..3 : 32-row slice
    const int wn   = wid >> 2;         // 0..1 : 64-col slice
    const int hi   = lane >> 4;        // lane group (0/1)
    const int ln   = lane & 15;        // position within 16

    const int m_blk = blockIdx.y * 128;
    const int n_blk = blockIdx.x * 128;

    // Issue async global->LDS copies for one 128x64 bf16 tile of A and W.
    // 1024 16-byte chunks per operand, 256 threads -> 4 chunks each.
    auto issue_tile = [&](int buf, int k0) {
#pragma unroll
        for (int c = tid; c < 128 * 8; c += 256) {
            const int row = c >> 3;
            const int off = (c & 7) * 8;
            const __bf16* ga = &A[(size_t)(m_blk + row) * K + k0 + off];
            const __bf16* gw = &W[(size_t)(n_blk + row) * K + k0 + off];
            unsigned la = (unsigned)(size_t)&As[buf][row * 64 + off];
            unsigned lb = (unsigned)(size_t)&Bs[buf][row * 64 + off];
            asm volatile("global_load_async_to_lds_b128 %0, %1, off"
                         :: "v"(la), "v"(ga) : "memory");
            asm volatile("global_load_async_to_lds_b128 %0, %1, off"
                         :: "v"(lb), "v"(gw) : "memory");
        }
    };

    v8f acc[2][4];
#pragma unroll
    for (int nt = 0; nt < 4; ++nt) {
        const int col = n_blk + wn * 64 + nt * 16 + ln;
        const float bv = bias ? bias[col] : 0.0f;
#pragma unroll
        for (int mt = 0; mt < 2; ++mt) {
#pragma unroll
            for (int r = 0; r < 8; ++r) acc[mt][nt][r] = bv;
        }
    }

    issue_tile(0, 0);

    int buf = 0;
    for (int k0 = 0; k0 < K; k0 += 64, buf ^= 1) {
        // My async loads for `buf` complete; barrier makes it true block-wide
        // and guarantees nobody still reads the buffer we're about to refill.
        asm volatile("s_wait_asynccnt 0x0" ::: "memory");
        __syncthreads();
        if (k0 + 64 < K) issue_tile(buf ^ 1, k0 + 64);

        const __bf16* Ab = As[buf];
        const __bf16* Bb = Bs[buf];
#pragma unroll
        for (int kk = 0; kk < 64; kk += 32) {
            // A fragment, 16-bit 16x32 layout:
            // lanes 0-15: M=lane, K = 0..7 then 16..23 ; lanes 16-31: K = 8..15 then 24..31
            v16bf afrag[2];
#pragma unroll
            for (int mt = 0; mt < 2; ++mt) {
                const int row = wm * 32 + mt * 16 + ln;
                const __bf16* p = &Ab[row * 64 + kk];
                v8bf lo = *(const v8bf*)(p + hi * 8);
                v8bf h8 = *(const v8bf*)(p + 16 + hi * 8);
                afrag[mt] = __builtin_shufflevector(lo, h8,
                    0, 1, 2, 3, 4, 5, 6, 7, 8, 9, 10, 11, 12, 13, 14, 15);
            }
            // B fragment, 16-bit 32x16 layout:
            // lanes 0-15: N=lane, K = 0..7 then 16..23 ; lanes 16-31: N=lane-16, K = 8..15 then 24..31
            v16bf bfrag[4];
#pragma unroll
            for (int nt = 0; nt < 4; ++nt) {
                const int col = wn * 64 + nt * 16 + ln;
                const __bf16* p = &Bb[col * 64 + kk];
                v8bf lo = *(const v8bf*)(p + hi * 8);
                v8bf h8 = *(const v8bf*)(p + 16 + hi * 8);
                bfrag[nt] = __builtin_shufflevector(lo, h8,
                    0, 1, 2, 3, 4, 5, 6, 7, 8, 9, 10, 11, 12, 13, 14, 15);
            }
#pragma unroll
            for (int mt = 0; mt < 2; ++mt)
#pragma unroll
                for (int nt = 0; nt < 4; ++nt)
                    acc[mt][nt] = __builtin_amdgcn_wmma_f32_16x16x32_bf16(
                        false, afrag[mt], false, bfrag[nt],
                        (short)0, acc[mt][nt], false, false);
        }
    }

    // C/D layout: VGPR r -> row (hi*8 + r), lane ln -> column.
#pragma unroll
    for (int mt = 0; mt < 2; ++mt) {
        const int rowbase = m_blk + wm * 32 + mt * 16 + hi * 8;
#pragma unroll
        for (int nt = 0; nt < 4; ++nt) {
            const int col = n_blk + wn * 64 + nt * 16 + ln;
#pragma unroll
            for (int r = 0; r < 8; ++r)
                C[(size_t)(rowbase + r) * N + col] = acc[mt][nt][r];
        }
    }
}

// ---------------------------------------------------------------------------
// Depthwise causal conv (k=4) + bias, gated by silu(z).
// xz: [b, L, 2*D_INNER] (first half = xh stream, second half = z)
// ---------------------------------------------------------------------------
__global__ void conv_silu_kernel(const float* __restrict__ xz,
                                 const float* __restrict__ conv_w,
                                 const float* __restrict__ conv_b,
                                 float* __restrict__ xh) {
    const int idx = blockIdx.x * blockDim.x + threadIdx.x;
    const int total = BATCH * SEQ_L * D_INNER;
    if (idx >= total) return;
    const int i  = idx % D_INNER;
    const int l  = (idx / D_INNER) % SEQ_L;
    const int bb = idx / (D_INNER * SEQ_L);

    const float* w = conv_w + i * D_CONV;
    float acc = conv_b[i];
#pragma unroll
    for (int t = 0; t < D_CONV; ++t) {
        const int ls = l - (D_CONV - 1) + t;
        if (ls >= 0)
            acc += xz[((size_t)(bb * SEQ_L + ls)) * (2 * D_INNER) + i] * w[t];
    }
    const float zv  = xz[((size_t)(bb * SEQ_L + l)) * (2 * D_INNER) + D_INNER + i];
    const float sig = 1.0f / (1.0f + __expf(-zv));
    xh[idx] = acc * (zv * sig);
}

// ---------------------------------------------------------------------------
// x_proj: proj[r, 0:33] = xh[r, :] . xproj_w[j, :]   (33 tiny dot products)
// One block (288 threads = 36 groups of 8) per row; group j reduces one output.
// ---------------------------------------------------------------------------
__global__ __launch_bounds__(288)
void xproj_kernel(const float* __restrict__ xh,
                  const float* __restrict__ xproj_w,
                  float* __restrict__ dlt,
                  float* __restrict__ Bc,
                  float* __restrict__ Cc) {
    __shared__ float red[288];
    const int r   = blockIdx.x;           // 0 .. M_ROWS-1
    const int tid = threadIdx.x;
    const int j   = tid >> 3;             // output index 0..35 (33 valid)
    const int sub = tid & 7;

    float acc = 0.0f;
    if (j < 2 * D_STATE + 1) {
        const float* xr = xh + (size_t)r * D_INNER;
        const float* wr = xproj_w + (size_t)j * D_INNER;
        for (int k = sub; k < D_INNER; k += 8) acc += xr[k] * wr[k];
    }
    red[tid] = acc;
    __syncthreads();
    if (sub == 0 && j < 2 * D_STATE + 1) {
        float s = 0.0f;
#pragma unroll
        for (int t = 0; t < 8; ++t) s += red[j * 8 + t];
        if (j == 0)                dlt[r] = s;
        else if (j <= D_STATE)     Bc[(size_t)r * D_STATE + (j - 1)] = s;
        else                       Cc[(size_t)r * D_STATE + (j - 1 - D_STATE)] = s;
    }
}

// ---------------------------------------------------------------------------
// Selective-scan recurrence. One thread per (batch, channel); 16 states in
// registers; B/C/dlt broadcast per step through LDS. exp(softplus(x)) == 1+e^x.
// Writes y directly as bf16 for the out-projection WMMA GEMM.
// ---------------------------------------------------------------------------
__global__ __launch_bounds__(256)
void ssm_scan_kernel(const float* __restrict__ xh,
                     const float* __restrict__ dlt,
                     const float* __restrict__ Bc,
                     const float* __restrict__ Cc,
                     const float* __restrict__ A_log,
                     const float* __restrict__ dt_w,
                     const float* __restrict__ dt_b,
                     const float* __restrict__ Dp,
                     __bf16* __restrict__ ybf) {
    const int bb = blockIdx.y;
    const int i  = blockIdx.x * 256 + threadIdx.x;

    float Areg[D_STATE];
#pragma unroll
    for (int s = 0; s < D_STATE; ++s)
        Areg[s] = -__expf(A_log[(size_t)i * D_STATE + s]);
    const float dtw = dt_w[i];
    const float dtb = dt_b[i];
    const float Dd  = Dp[i];

    float h[D_STATE];
#pragma unroll
    for (int s = 0; s < D_STATE; ++s) h[s] = 0.0f;

    __shared__ float sB[D_STATE];
    __shared__ float sC[D_STATE];
    __shared__ float sDlt;

    for (int l = 0; l < SEQ_L; ++l) {
        const size_t r = (size_t)bb * SEQ_L + l;
        __syncthreads();
        if (threadIdx.x < D_STATE)            sB[threadIdx.x] = Bc[r * D_STATE + threadIdx.x];
        else if (threadIdx.x < 2 * D_STATE)   sC[threadIdx.x - D_STATE] = Cc[r * D_STATE + threadIdx.x - D_STATE];
        else if (threadIdx.x == 2 * D_STATE)  sDlt = dlt[r];
        __syncthreads();

        const float pre   = sDlt * dtw + dtb;
        const float delta = 1.0f + __expf(pre);      // exp(softplus(pre))
        const float xv    = xh[r * D_INNER + i];

        float acc = 0.0f;
#pragma unroll
        for (int s = 0; s < D_STATE; ++s) {
            const float ab = __expf(Areg[s] * delta);
            h[s] = ab * h[s] + (delta * sB[s]) * xv;
            acc += h[s] * sC[s];
        }
        ybf[r * D_INNER + i] = (__bf16)(acc + Dd * xv);
    }
}

// ---------------------------------------------------------------------------
// Launch
// ---------------------------------------------------------------------------
extern "C" void kernel_launch(void* const* d_in, const int* in_sizes, int n_in,
                              void* d_out, int out_size, void* d_ws, size_t ws_size,
                              hipStream_t stream) {
    (void)in_sizes; (void)n_in; (void)out_size; (void)ws_size;

    const float* x       = (const float*)d_in[0];   // [2,1024,1024]
    const float* in_w    = (const float*)d_in[1];   // [4096,1024]
    const float* in_b    = (const float*)d_in[2];   // [4096]
    const float* conv_w  = (const float*)d_in[3];   // [2048,1,4]
    const float* conv_b  = (const float*)d_in[4];   // [2048]
    const float* xproj_w = (const float*)d_in[5];   // [33,2048]
    const float* dt_w    = (const float*)d_in[6];   // [2048,1]
    const float* dt_b    = (const float*)d_in[7];   // [2048]
    const float* A_log   = (const float*)d_in[8];   // [2048,16]
    const float* Dp      = (const float*)d_in[9];   // [2048]
    const float* out_w   = (const float*)d_in[10];  // [1024,2048]
    float* out           = (float*)d_out;           // [2,1024,1024]

    // Workspace carve-up (256B aligned)
    char* ws = (char*)d_ws;
    size_t off = 0;
    auto carve = [&](size_t bytes) -> char* {
        char* p = ws + off;
        off += (bytes + 255) & ~(size_t)255;
        return p;
    };
    __bf16* bx     = (__bf16*)carve((size_t)M_ROWS * D_MODEL * 2);        // 4 MB
    __bf16* bw_in  = (__bf16*)carve((size_t)2 * D_INNER * D_MODEL * 2);   // 8 MB
    __bf16* bw_out = (__bf16*)carve((size_t)D_MODEL * D_INNER * 2);       // 4 MB
    float*  xz     = (float*) carve((size_t)M_ROWS * 2 * D_INNER * 4);    // 32 MB
    float*  xh     = (float*) carve((size_t)M_ROWS * D_INNER * 4);        // 16 MB
    float*  dltb   = (float*) carve((size_t)M_ROWS * 4);
    float*  Bc     = (float*) carve((size_t)M_ROWS * D_STATE * 4);
    float*  Cc     = (float*) carve((size_t)M_ROWS * D_STATE * 4);
    __bf16* ybf    = (__bf16*)carve((size_t)M_ROWS * D_INNER * 2);        // 8 MB

    // 1) bf16 staging of GEMM operands
    {
        int n;
        n = M_ROWS * D_MODEL;
        f32_to_bf16_kernel<<<(n + 255) / 256, 256, 0, stream>>>(x, bx, n);
        n = 2 * D_INNER * D_MODEL;
        f32_to_bf16_kernel<<<(n + 255) / 256, 256, 0, stream>>>(in_w, bw_in, n);
        n = D_MODEL * D_INNER;
        f32_to_bf16_kernel<<<(n + 255) / 256, 256, 0, stream>>>(out_w, bw_out, n);
    }

    // 2) in-projection: xz[2048, 4096] = x @ in_w^T + in_b
    wmma_gemm_bf16<<<dim3((2 * D_INNER) / 128, M_ROWS / 128), 256, 0, stream>>>(
        bx, bw_in, in_b, xz, M_ROWS, 2 * D_INNER, D_MODEL);

    // 3) depthwise causal conv + silu gate -> xh
    {
        int n = BATCH * SEQ_L * D_INNER;
        conv_silu_kernel<<<(n + 255) / 256, 256, 0, stream>>>(xz, conv_w, conv_b, xh);
    }

    // 4) x_proj -> dlt / B / C
    xproj_kernel<<<M_ROWS, 288, 0, stream>>>(xh, xproj_w, dltb, Bc, Cc);

    // 5) selective scan -> y (bf16)
    ssm_scan_kernel<<<dim3(D_INNER / 256, BATCH), 256, 0, stream>>>(
        xh, dltb, Bc, Cc, A_log, dt_w, dt_b, Dp, ybf);

    // 6) out-projection: out[2048, 1024] = y @ out_w^T
    wmma_gemm_bf16<<<dim3(D_MODEL / 128, M_ROWS / 128), 256, 0, stream>>>(
        ybf, bw_out, nullptr, out, M_ROWS, D_MODEL, D_INNER);
}
